// Transformer_5712306504259
// MI455X (gfx1250) — matrix operand order repaired and checked by reading, not compiled, not run on hardware
//
#include <hip/hip_runtime.h>
#include <math.h>

// ---------------------------------------------------------------------------
// Problem constants (from the reference)
// ---------------------------------------------------------------------------
#define B_    4
#define N_    128
#define M_    129          // N + dummy atom
#define MM_   (M_ * M_)    // 16641
#define AD_   512
#define BD_   256
#define NH_   16
#define RH_   64
#define NL_   6
#define AF_   ((NL_ + 1) * AD_)   // 3584
#define BF_   ((NL_ + 1) * BD_)   // 1792
#define BM_   (B_ * M_)           // 516
#define BMM_  (B_ * MM_)          // 66564
#define LN_EPS 1e-3f

typedef __attribute__((ext_vector_type(16))) _Float16 v16h;
typedef __attribute__((ext_vector_type(8)))  _Float16 h8;
typedef __attribute__((ext_vector_type(8)))  float    v8f;

#define SHUF16(a, b) __builtin_shufflevector(a, b, \
    0,1,2,3,4,5,6,7,8,9,10,11,12,13,14,15)

// ---------------------------------------------------------------------------
// Generic batched WMMA GEMM:
//   C = act( scale * (A @ B) + bias + addsrc )
// A: fp32, [Mrows x K] per batch, row-major, lda; per-batch offset a_bs.
// B: fp32; if b_is_nk: B is [N x K] ("Bt"); else [K x N], ldb.
// C: fp32 generalized scatter:
//   outer = batch / c_inner, inner = batch % c_inner
//   C[outer*c_so + inner*c_si + row*c_sr + col*c_sc]
// Tiling: 128x64 block tile, 8 waves (wave32), each wave -> 32x32 (4 accs),
// K-step 32.  Core: v_wmma_f32_16x16x32_f16, fp32 staged to f16 via LDS.
// Fast path (block-uniform full tile): vectorized/unguarded loads;
// edge path: clamped addresses + v_cndmask select (no exec-mask branching).
// ---------------------------------------------------------------------------
__global__ __launch_bounds__(256)
void k_gemm(const float* __restrict__ A, long long a_bs, int lda, int a_v4,
            const float* __restrict__ Bm, long long b_bs, int ldb, int b_is_nk,
            int b_v4,
            const float* __restrict__ bias,
            const float* __restrict__ addsrc, long long add_bs, int ld_add,
            float* __restrict__ C,
            long long c_so, long long c_si, int c_inner,
            long long c_sr, long long c_sc,
            int Mrows, int Ncols, int K, float scale, int relu)
{
    __shared__ _Float16 As[128][32];
    __shared__ _Float16 Bs[64][32];   // transposed tile: [n][k]

    const int tid  = threadIdx.x;
    const int lane = tid & 31;
    const int wave = tid >> 5;
    const int wy   = wave >> 1;       // 0..3  (M direction, 32 rows each)
    const int wx   = wave & 1;        // 0..1  (N direction, 32 cols each)
    const int lh   = lane & 15;
    const int lg   = lane >> 4;

    const int batch = blockIdx.z;
    const long long aoff = (long long)batch * a_bs;
    const long long boff = (long long)batch * b_bs;
    const int outer = (c_inner > 1) ? (batch / c_inner) : batch;
    const int inner = (c_inner > 1) ? (batch % c_inner) : 0;
    const long long coff = (long long)outer * c_so + (long long)inner * c_si;

    const int mBase = blockIdx.x * 128;
    const int nBase = blockIdx.y * 64;
    const bool fullM = (mBase + 128 <= Mrows);
    const bool fullN = (nBase + 64  <= Ncols);

    // cooperative-load indices: thread t handles 8 contiguous k of one row
    const int t8 = tid * 8;
    const int lr = t8 >> 5;           // 0..63
    const int lc = t8 & 31;           // 0,8,16,24

    v8f acc[2][2] = {{{}, {}}, {{}, {}}};

    for (int k0 = 0; k0 < K; k0 += 32) {
        const bool fullK = (k0 + 32 <= K);
        // ---- stage A tile (128 x 32), two 64-row halves per thread ----
#pragma unroll
        for (int rep = 0; rep < 2; ++rep) {
            const int r    = lr + rep * 64;
            const int grow = mBase + r;
            h8 v = {};
            if (fullM && fullK) {
                const float* ap = A + aoff + (long long)grow * lda + (k0 + lc);
                if (k0 + 32 < K) __builtin_prefetch(ap + 32, 0, 0);
                if (a_v4) {
                    const float4 f0 = *(const float4*)ap;
                    const float4 f1 = *(const float4*)(ap + 4);
                    v[0] = (_Float16)f0.x; v[1] = (_Float16)f0.y;
                    v[2] = (_Float16)f0.z; v[3] = (_Float16)f0.w;
                    v[4] = (_Float16)f1.x; v[5] = (_Float16)f1.y;
                    v[6] = (_Float16)f1.z; v[7] = (_Float16)f1.w;
                } else {
#pragma unroll
                    for (int i = 0; i < 8; ++i) v[i] = (_Float16)ap[i];
                }
            } else {
                // branch-free edge path: clamp address, select value
                const int gr = (grow < Mrows) ? grow : (Mrows - 1);
                const float* ap = A + aoff + (long long)gr * lda;
#pragma unroll
                for (int i = 0; i < 8; ++i) {
                    const int kk = k0 + lc + i;
                    const int kc = (kk < K) ? kk : (K - 1);
                    const float val = ap[kc];
                    v[i] = (_Float16)((grow < Mrows && kk < K) ? val : 0.f);
                }
            }
            *(h8*)(&As[r][lc]) = v;
        }
        // ---- stage B tile as [n][k] (64 x 32) ----
        {
            h8 v = {};
            const int gn = nBase + lr;
            if (fullN && fullK) {
                if (b_is_nk) {
                    const float* bp = Bm + boff + (long long)gn * ldb + (k0 + lc);
                    if (k0 + 32 < K) __builtin_prefetch(bp + 32, 0, 0);
                    if (b_v4) {
                        const float4 f0 = *(const float4*)bp;
                        const float4 f1 = *(const float4*)(bp + 4);
                        v[0] = (_Float16)f0.x; v[1] = (_Float16)f0.y;
                        v[2] = (_Float16)f0.z; v[3] = (_Float16)f0.w;
                        v[4] = (_Float16)f1.x; v[5] = (_Float16)f1.y;
                        v[6] = (_Float16)f1.z; v[7] = (_Float16)f1.w;
                    } else {
#pragma unroll
                        for (int i = 0; i < 8; ++i) v[i] = (_Float16)bp[i];
                    }
                } else {
                    const float* bp = Bm + boff + (long long)(k0 + lc) * ldb + gn;
#pragma unroll
                    for (int i = 0; i < 8; ++i)
                        v[i] = (_Float16)bp[(long long)i * ldb];
                }
            } else {
                const int gc = (gn < Ncols) ? gn : (Ncols - 1);
#pragma unroll
                for (int i = 0; i < 8; ++i) {
                    const int kk = k0 + lc + i;
                    const int kc = (kk < K) ? kk : (K - 1);
                    const float val = b_is_nk
                        ? Bm[boff + (long long)gc * ldb + kc]
                        : Bm[boff + (long long)kc * ldb + gc];
                    v[i] = (_Float16)((gn < Ncols && kk < K) ? val : 0.f);
                }
            }
            *(h8*)(&Bs[lr][lc]) = v;
        }
        __syncthreads();

        // ---- fragments per ISA 7.12.2 wave32 layouts ----
        v16h af[2], bf[2];
#pragma unroll
        for (int mt = 0; mt < 2; ++mt) {
            const int row = wy * 32 + mt * 16 + lh;
            h8 a0 = *(const h8*)(&As[row][lg * 8]);
            h8 a1 = *(const h8*)(&As[row][16 + lg * 8]);
            af[mt] = SHUF16(a0, a1);
        }
#pragma unroll
        for (int nt = 0; nt < 2; ++nt) {
            const int col = wx * 32 + nt * 16 + lh;
            h8 b0 = *(const h8*)(&Bs[col][lg * 8]);
            h8 b1 = *(const h8*)(&Bs[col][16 + lg * 8]);
            bf[nt] = SHUF16(b0, b1);
        }
#pragma unroll
        for (int mt = 0; mt < 2; ++mt)
#pragma unroll
            for (int nt = 0; nt < 2; ++nt)
                acc[mt][nt] = __builtin_amdgcn_wmma_f32_16x16x32_f16(
                    false, af[mt], false, bf[nt], (short)0,
                    acc[mt][nt], false, false);
        __syncthreads();
    }

    // ---- epilogue: C/D layout: lanes 0-15 -> M=r, lanes 16-31 -> M=8+r ----
#pragma unroll
    for (int mt = 0; mt < 2; ++mt) {
        const int rbase = mBase + wy * 32 + mt * 16 + ((lane < 16) ? 0 : 8);
#pragma unroll
        for (int nt = 0; nt < 2; ++nt) {
            const int cb = nBase + wx * 32 + nt * 16 + lh;
            if (cb >= Ncols) continue;
#pragma unroll
            for (int r = 0; r < 8; ++r) {
                const int gm = rbase + r;
                if (gm >= Mrows) continue;
                float v = acc[mt][nt][r] * scale;
                if (bias)   v += bias[cb];
                if (addsrc) v += addsrc[(long long)batch * add_bs +
                                        (long long)gm * ld_add + cb];
                if (relu && v < 0.f) v = 0.f;
                C[coff + (long long)gm * c_sr + (long long)cb * c_sc] = v;
            }
        }
    }
}

// ---------------------------------------------------------------------------
// Residual + LayerNorm: out[row] = LN(x[row] + res[row]) * g + b
// one wave32 per row; D in {256, 512}
// ---------------------------------------------------------------------------
__global__ __launch_bounds__(256)
void k_ln(const float* __restrict__ x, int ldx,
          const float* __restrict__ res, int ldres,
          const float* __restrict__ g, const float* __restrict__ beta,
          float* __restrict__ out, int ldout, int rows, int D)
{
    const int wave = threadIdx.x >> 5, lane = threadIdx.x & 31;
    const int row = blockIdx.x * 8 + wave;
    if (row >= rows) return;
    const float* xp = x   + (long long)row * ldx;
    const float* rp = res + (long long)row * ldres;
    float hv[16];
    const int n = D >> 5;
    float s = 0.f, ss = 0.f;
    for (int i = 0; i < n; ++i) {
        const float h = xp[lane + i * 32] + rp[lane + i * 32];
        hv[i] = h; s += h; ss += h * h;
    }
    for (int o = 16; o; o >>= 1) {
        s  += __shfl_xor(s,  o, 32);
        ss += __shfl_xor(ss, o, 32);
    }
    const float mu  = s / (float)D;
    const float var = ss / (float)D - mu * mu;
    const float rstd = rsqrtf(var + LN_EPS);
    float* op = out + (long long)row * ldout;
    for (int i = 0; i < n; ++i) {
        const int c = lane + i * 32;
        op[c] = (hv[i] - mu) * rstd * g[c] + beta[c];
    }
}

// ---------------------------------------------------------------------------
// In-place softmax over rows of length `len` (len = M_ = 129); wave32 per row
// ---------------------------------------------------------------------------
__global__ __launch_bounds__(256)
void k_softmax(float* __restrict__ x, int len, long long rows)
{
    const int wave = threadIdx.x >> 5, lane = threadIdx.x & 31;
    const long long row = (long long)blockIdx.x * 8 + wave;
    if (row >= rows) return;
    float* p = x + row * (long long)len;
    float hv[8];
    const int n = (len + 31) >> 5;
    float mx = -1e30f;
    for (int i = 0; i < n; ++i) {
        const int c = lane + i * 32;
        const float v = (c < len) ? p[c] : -1e30f;
        hv[i] = v; mx = fmaxf(mx, v);
    }
    for (int o = 16; o; o >>= 1) mx = fmaxf(mx, __shfl_xor(mx, o, 32));
    float s = 0.f;
    for (int i = 0; i < n; ++i) {
        const int c = lane + i * 32;
        const float e = __expf(hv[i] - mx);
        hv[i] = e;
        if (c < len) s += e;
    }
    for (int o = 16; o; o >>= 1) s += __shfl_xor(s, o, 32);
    const float inv = 1.f / s;
    for (int i = 0; i < n; ++i) {
        const int c = lane + i * 32;
        if (c < len) p[c] = hv[i] * inv;
    }
}

// ---------------------------------------------------------------------------
// Input embeddings (K = 5 / 4: too small for WMMA; bandwidth-trivial)
// ---------------------------------------------------------------------------
__global__ void k_prep_atoms(const float* __restrict__ atoms,
                             const float* __restrict__ dummy,
                             const float* __restrict__ W,
                             const float* __restrict__ bias,
                             float* __restrict__ out, int ldout)
{
    const long long idx = (long long)blockIdx.x * blockDim.x + threadIdx.x;
    const long long total = (long long)BM_ * AD_;
    if (idx >= total) return;
    const int col = (int)(idx % AD_);
    const int row = (int)(idx / AD_);
    const int b = row / M_, m = row % M_;
    float f[5];
    if (m < N_) {
#pragma unroll
        for (int c = 0; c < 5; ++c)
            f[c] = atoms[((long long)b * N_ + m) * 5 + c] * 2.f - 1.f;
    } else {
#pragma unroll
        for (int c = 0; c < 5; ++c) f[c] = dummy[c];
    }
    float acc = bias[col];
#pragma unroll
    for (int c = 0; c < 5; ++c) acc += f[c] * W[c * AD_ + col];
    out[(long long)row * ldout + col] = acc;
}

__global__ void k_prep_bonds(const float* __restrict__ bonds,
                             const float* __restrict__ W,
                             const float* __restrict__ bias,
                             float* __restrict__ out, int ldout)
{
    const long long idx = (long long)blockIdx.x * blockDim.x + threadIdx.x;
    const long long total = (long long)BMM_ * BD_;
    if (idx >= total) return;
    const int col = (int)(idx % BD_);
    long long r = idx / BD_;
    const int b  = (int)(r / MM_);
    const int ij = (int)(r % MM_);
    const int i = ij / M_, j = ij % M_;
    float f0, f1, f2, f3;
    if (i < N_ && j < N_) {
        const float msk = (i == j) ? 0.f : 1.f;
        const float* bp = bonds + (((long long)b * N_ + i) * N_ + j) * 3;
        f0 = bp[0] * msk * 2.f - 1.f;
        f1 = bp[1] * msk * 2.f - 1.f;
        f2 = bp[2] * msk * 2.f - 1.f;
        f3 = 0.f;                       // padded after *2-1 -> stays 0
    } else {
        f0 = f1 = f2 = 0.f;             // zero-padded region
        f3 = 1.f;                       // dummy channel
    }
    float acc = bias[col];
    acc += f0 * W[0 * BD_ + col] + f1 * W[1 * BD_ + col]
         + f2 * W[2 * BD_ + col] + f3 * W[3 * BD_ + col];
    out[r * ldout + col] = acc;
}

// ---------------------------------------------------------------------------
// Gather final outputs: drop dummy atom row/col, apply diag mask on bonds
// ---------------------------------------------------------------------------
#define A_OUT_ (B_ * N_ * 5)
#define B_OUT_ (B_ * N_ * N_ * 3)
__global__ void k_final(const float* __restrict__ afin,
                        const float* __restrict__ bfin,
                        float* __restrict__ out)
{
    const long long idx = (long long)blockIdx.x * blockDim.x + threadIdx.x;
    if (idx < A_OUT_) {
        const int c = (int)(idx % 5);
        const int r = (int)(idx / 5);
        const int b = r / N_, n = r % N_;
        out[idx] = afin[((long long)b * M_ + n) * 5 + c];
    } else if (idx < (long long)A_OUT_ + B_OUT_) {
        const long long k = idx - A_OUT_;
        const int c = (int)(k % 3);
        long long t = k / 3;
        const int j = (int)(t % N_); t /= N_;
        const int i = (int)(t % N_);
        const int b = (int)(t / N_);
        const float v = bfin[(((long long)b * M_ + i) * M_ + j) * 3 + c];
        out[idx] = (i == j) ? 0.f : v;
    }
}

// ---------------------------------------------------------------------------
// Host-side helpers
// ---------------------------------------------------------------------------
static float* wsAlloc(char** cur, size_t elems)
{
    float* p = (float*)*cur;
    *cur += (elems * sizeof(float) + 255) & ~(size_t)255;
    return p;
}

static void gemm(hipStream_t s,
                 const float* A, long long a_bs, int lda,
                 const float* Bmat, long long b_bs, int ldb, int b_nk,
                 const float* bias, const float* add, long long add_bs, int ld_add,
                 float* C, long long c_so, long long c_si, int c_inner,
                 long long c_sr, long long c_sc,
                 int Mr, int Nc, int K, float scale, int relu, int batch)
{
    const int a_v4 = ((lda & 3) == 0) && ((a_bs & 3) == 0);
    const int b_v4 = b_nk && ((ldb & 3) == 0) && ((b_bs & 3) == 0);
    dim3 g((Mr + 127) / 128, (Nc + 63) / 64, batch);
    k_gemm<<<g, 256, 0, s>>>(A, a_bs, lda, a_v4, Bmat, b_bs, ldb, b_nk, b_v4,
                             bias, add, add_bs, ld_add,
                             C, c_so, c_si, c_inner, c_sr, c_sc,
                             Mr, Nc, K, scale, relu);
}

// standard dense GEMM: C[Mr x Nc] = act(A @ W + bias), W is [K x Nc]
static void gemmStd(hipStream_t s, const float* A, int lda,
                    const float* W, const float* bias,
                    float* C, int ldc, int Mr, int Nc, int K, int relu)
{
    gemm(s, A, 0, lda, W, 0, Nc, /*b_nk=*/0, bias, nullptr, 0, 0,
         C, 0, 0, 1, ldc, 1, Mr, Nc, K, 1.f, relu, 1);
}

static void lnRes(hipStream_t s, const float* x, int ldx,
                  const float* res, int ldres,
                  const float* g, const float* b,
                  float* out, int ldout, int rows, int D)
{
    k_ln<<<dim3((rows + 7) / 8), 256, 0, s>>>(x, ldx, res, ldres, g, b,
                                              out, ldout, rows, D);
}

// ---------------------------------------------------------------------------
// kernel_launch
// ---------------------------------------------------------------------------
extern "C" void kernel_launch(void* const* d_in, const int* in_sizes, int n_in,
                              void* d_out, int out_size, void* d_ws, size_t ws_size,
                              hipStream_t stream)
{
    (void)in_sizes; (void)n_in; (void)out_size; (void)ws_size;

    // --- inputs (setup_inputs dict order, params flattened in insertion order)
    const float* in_atoms  = (const float*)d_in[0];
    const float* in_bonds  = (const float*)d_in[1];
    const float* dummyA    = (const float*)d_in[2];
    const float* in_atom_w = (const float*)d_in[3];
    const float* in_atom_b = (const float*)d_in[4];
    const float* in_bond_w = (const float*)d_in[5];
    const float* in_bond_b = (const float*)d_in[6];
#define BLKP(l, j) ((const float*)d_in[7 + (l) * 25 + (j)])
    // j: 0 wq 1 wk 2 wv 3 wcond 4 att_ow 5 att_ob 6 ln_a1_g 7 ln_a1_b
    //    8 rec_wq 9 rec_ow 10 rec_ob 11 ln_b1_g 12 ln_b1_b
    //    13 affn_w1 14 affn_b1 15 affn_w2 16 affn_b2 17 ln_a2_g 18 ln_a2_b
    //    19 bffn_w1 20 bffn_b1 21 bffn_w2 22 bffn_b2 23 ln_b2_g 24 ln_b2_b
    const int OB = 7 + NL_ * 25;
    const float* out_a_w1 = (const float*)d_in[OB + 0];
    const float* out_a_b1 = (const float*)d_in[OB + 1];
    const float* out_a_w2 = (const float*)d_in[OB + 2];
    const float* out_a_b2 = (const float*)d_in[OB + 3];
    const float* out_a_w3 = (const float*)d_in[OB + 4];
    const float* out_a_b3 = (const float*)d_in[OB + 5];
    const float* out_a_w  = (const float*)d_in[OB + 6];
    const float* out_a_b  = (const float*)d_in[OB + 7];
    const float* out_b_w1 = (const float*)d_in[OB + 8];
    const float* out_b_b1 = (const float*)d_in[OB + 9];
    const float* out_b_w2 = (const float*)d_in[OB + 10];
    const float* out_b_b2 = (const float*)d_in[OB + 11];
    const float* out_b_w3 = (const float*)d_in[OB + 12];
    const float* out_b_b3 = (const float*)d_in[OB + 13];
    const float* out_b_w  = (const float*)d_in[OB + 14];
    const float* out_b_b  = (const float*)d_in[OB + 15];

    // --- workspace layout
    char* cur = (char*)d_ws;
    float* aconcat = wsAlloc(&cur, (size_t)BM_ * AF_);    // [BM, 7*AD]
    float* bconcat = wsAlloc(&cur, (size_t)BMM_ * BF_);   // [BMM, 7*BD]
    float* qbuf    = wsAlloc(&cur, (size_t)BM_ * AD_);
    float* kbuf    = wsAlloc(&cur, (size_t)BM_ * AD_);
    float* vbuf    = wsAlloc(&cur, (size_t)BM_ * AD_);
    float* cond    = wsAlloc(&cur, (size_t)B_ * NH_ * MM_);
    float* scores  = wsAlloc(&cur, (size_t)B_ * NH_ * MM_);
    float* rbuf    = wsAlloc(&cur, (size_t)BM_ * AD_);
    float* ta1     = wsAlloc(&cur, (size_t)BM_ * AD_);
    float* ta2     = wsAlloc(&cur, (size_t)BM_ * AD_);
    float* amid    = wsAlloc(&cur, (size_t)BM_ * AD_);
    float* rqbuf   = wsAlloc(&cur, (size_t)BM_ * BD_);
    float* ratt    = wsAlloc(&cur, (size_t)BMM_ * RH_);
    float* tb1     = wsAlloc(&cur, (size_t)BMM_ * BD_);
    float* tb2     = wsAlloc(&cur, (size_t)BMM_ * BD_);
    float* bmid    = wsAlloc(&cur, (size_t)BMM_ * BD_);
    float* afin    = wsAlloc(&cur, (size_t)BM_ * 8);
    float* bfin    = wsAlloc(&cur, (size_t)BMM_ * 4);

    const float inv_dk_a = 1.f / sqrtf((float)(AD_ / NH_));  // 1/sqrt(32)
    const float inv_dk_r = 1.f / sqrtf((float)(BD_ / RH_));  // 1/2

    // --- input embeddings -> concat slice 0
    {
        const long long ta = (long long)BM_ * AD_;
        k_prep_atoms<<<dim3((unsigned)((ta + 255) / 256)), 256, 0, stream>>>(
            in_atoms, dummyA, in_atom_w, in_atom_b, aconcat, AF_);
        const long long tb = (long long)BMM_ * BD_;
        k_prep_bonds<<<dim3((unsigned)((tb + 255) / 256)), 256, 0, stream>>>(
            in_bonds, in_bond_w, in_bond_b, bconcat, BF_);
    }

    // --- transformer blocks
    for (int l = 0; l < NL_; ++l) {
        const float* aL = aconcat + (long long)l * AD_;   // lda = AF_
        const float* bL = bconcat + (long long)l * BD_;   // lda = BF_

        // q, k, v = atoms @ W  (no bias)
        gemmStd(stream, aL, AF_, BLKP(l, 0), nullptr, qbuf, AD_, BM_, AD_, AD_, 0);
        gemmStd(stream, aL, AF_, BLKP(l, 1), nullptr, kbuf, AD_, BM_, AD_, AD_, 0);
        gemmStd(stream, aL, AF_, BLKP(l, 2), nullptr, vbuf, AD_, BM_, AD_, AD_, 0);

        // cond[b,h,i,j] = bonds @ wcond, scattered to [B,NH,M,M]
        gemm(stream, bL, (long long)MM_ * BF_, BF_,
             BLKP(l, 3), 0, NH_, /*b_nk=*/0,
             nullptr, nullptr, 0, 0,
             cond, (long long)NH_ * MM_, 0, 1, /*c_sr=*/1, /*c_sc=*/MM_,
             MM_, NH_, BD_, 1.f, 0, B_);

        // scores = q @ k^T / dk + cond   (batched over B*NH, raw-reshape heads)
        gemm(stream, qbuf, (long long)M_ * 32, 32,
             kbuf, (long long)M_ * 32, 32, /*b_nk=*/1,
             nullptr, cond, (long long)MM_, M_,
             scores, (long long)MM_, 0, 1, M_, 1,
             M_, M_, 32, inv_dk_a, 0, B_ * NH_);

        // softmax over last dim
        k_softmax<<<dim3((B_ * NH_ * M_ + 7) / 8), 256, 0, stream>>>(
            scores, M_, (long long)B_ * NH_ * M_);

        // r = att @ v  -> stored contiguous [B,NH,M,32] == raw-reshape [B,M,AD]
        gemm(stream, scores, (long long)MM_, M_,
             vbuf, (long long)M_ * 32, 32, /*b_nk=*/0,
             nullptr, nullptr, 0, 0,
             rbuf, (long long)M_ * 32, 0, 1, 32, 1,
             M_, 32, M_, 1.f, 0, B_ * NH_);

        // attention out proj + residual LN -> amid
        gemmStd(stream, rbuf, AD_, BLKP(l, 4), BLKP(l, 5), ta1, AD_, BM_, AD_, AD_, 0);
        lnRes(stream, ta1, AD_, aL, AF_, BLKP(l, 6), BLKP(l, 7), amid, AD_, BM_, AD_);

        // ReconnectAtoms: rq = amid @ rec_wq
        gemmStd(stream, amid, AD_, BLKP(l, 8), nullptr, rqbuf, BD_, BM_, BD_, AD_, 0);
        // ratt[b,h,i,j] = rq @ rq^T / dk_r, scattered to [B,M,M,RH]
        gemm(stream, rqbuf, (long long)M_ * 4, 4,
             rqbuf, (long long)M_ * 4, 4, /*b_nk=*/1,
             nullptr, nullptr, 0, 0,
             ratt, (long long)MM_ * RH_, 1, RH_,
             (long long)M_ * RH_, RH_,
             M_, M_, 4, inv_dk_r, 0, B_ * RH_);
        // bonds delta = ratt @ rec_ow + rec_ob ; residual LN -> bmid
        gemmStd(stream, ratt, RH_, BLKP(l, 9), BLKP(l, 10), tb1, BD_, BMM_, BD_, RH_, 0);
        lnRes(stream, tb1, BD_, bL, BF_, BLKP(l, 11), BLKP(l, 12), bmid, BD_, BMM_, BD_);

        // atom FFN -> concat slice l+1
        gemmStd(stream, amid, AD_, BLKP(l, 13), BLKP(l, 14), ta1, AD_, BM_, AD_, AD_, 1);
        gemmStd(stream, ta1, AD_, BLKP(l, 15), BLKP(l, 16), ta2, AD_, BM_, AD_, AD_, 0);
        lnRes(stream, ta2, AD_, amid, AD_, BLKP(l, 17), BLKP(l, 18),
              aconcat + (long long)(l + 1) * AD_, AF_, BM_, AD_);

        // bond FFN -> concat slice l+1   (dominant FLOPs)
        gemmStd(stream, bmid, BD_, BLKP(l, 19), BLKP(l, 20), tb1, BD_, BMM_, BD_, BD_, 1);
        gemmStd(stream, tb1, BD_, BLKP(l, 21), BLKP(l, 22), tb2, BD_, BMM_, BD_, BD_, 0);
        lnRes(stream, tb2, BD_, bmid, BD_, BLKP(l, 23), BLKP(l, 24),
              bconcat + (long long)(l + 1) * BD_, BF_, BMM_, BD_);
    }

    // --- output block (runs over all rows incl. dummy; gather drops it)
    gemmStd(stream, aconcat, AF_, out_a_w1, out_a_b1, ta1, AD_, BM_, AD_, AF_, 1);
    gemmStd(stream, ta1, AD_, out_a_w2, out_a_b2, ta2, AD_, BM_, AD_, AD_, 1);
    gemmStd(stream, ta2, AD_, out_a_w3, out_a_b3, ta1, AD_, BM_, AD_, AD_, 0);
    gemmStd(stream, ta1, AD_, out_a_w,  out_a_b,  afin, 5,  BM_, 5,   AD_, 0);

    gemmStd(stream, bconcat, BF_, out_b_w1, out_b_b1, tb1, BD_, BMM_, BD_, BF_, 1);
    gemmStd(stream, tb1, BD_, out_b_w2, out_b_b2, tb2, BD_, BMM_, BD_, BD_, 1);
    gemmStd(stream, tb2, BD_, out_b_w3, out_b_b3, tb1, BD_, BMM_, BD_, BD_, 0);
    gemmStd(stream, tb1, BD_, out_b_w,  out_b_b,  bfin, 3,  BMM_, 3,  BD_, 0);

    const long long tot = (long long)A_OUT_ + B_OUT_;
    k_final<<<dim3((unsigned)((tot + 255) / 256)), 256, 0, stream>>>(
        afin, bfin, (float*)d_out);
}
